// Seq2Seq_13254269076087
// MI455X (gfx1250) — matrix operand order repaired and checked by reading, not compiled, not run on hardware
//
#include <hip/hip_runtime.h>

// ---- problem constants (from reference) ----
#define BB 128   // batch
#define SS 512   // encoder sequence length
#define TT 128   // decoder target length
#define FF 64    // encoder input features
#define HH 1024  // hidden
#define G4 4096  // 4*H
#define KT_H 32  // H/32 k-tiles
#define KT_X 2   // F/32 k-tiles

typedef __bf16 bf16_t;
typedef __attribute__((ext_vector_type(16))) __bf16 bf16x16;
typedef __attribute__((ext_vector_type(8)))  float  v8f;

// ---- WMMA A-operand (16-bit, 16x32) lane/half mapping (ISA 7.12.2) ----
// lanes 0-15 : row = lane,    halves 0..7 -> K 0..7,  halves 8..15 -> K 16..23
// lanes 16-31: row = lane-16, halves 0..7 -> K 8..15, halves 8..15 -> K 24..31
__device__ __forceinline__ void amap(int row, int kk, int& lane, int& h) {
  if (kk < 8)       { lane = row;      h = kk;       }
  else if (kk < 16) { lane = row + 16; h = kk - 8;   }
  else if (kk < 24) { lane = row;      h = kk - 8;   }
  else              { lane = row + 16; h = kk - 16;  }
}

__device__ __forceinline__ float sigmf(float x) { return 1.0f / (1.0f + __expf(-x)); }

// ================= pack kernels (one-time per launch, fully parallel) ========

// W is (4096 x K) row-major f32. Pack as WMMA B fragments:
// out[((ntile*ktiles + ktile)*32 + lane)*16 + h], B-map:
//   lanes 0-15 : col = lane,    h 0..7 -> K 0..7,  h 8..15 -> K 16..23
//   lanes 16-31: col = lane-16, h 0..7 -> K 8..15, h 8..15 -> K 24..31
__global__ void __launch_bounds__(256)
pack_B(const float* __restrict__ W, bf16_t* __restrict__ out, int K, int ktiles) {
  int e = blockIdx.x * blockDim.x + threadIdx.x;
  if (e >= G4 * K) return;
  int j = e / K, k = e - j * K;
  int ntile = j >> 4, c16 = j & 15;
  int ktile = k >> 5, kk = k & 31;
  int lane = ((kk >> 3) & 1) ? (16 + c16) : c16;
  int h = (kk & 7) + ((kk >> 4) << 3);
  out[(((size_t)ntile * ktiles + ktile) * 32 + lane) * 16 + h] = (bf16_t)W[e];
}

// source (B,S,F) f32 -> per-step A fragments: step t block of 8192 halves:
// [mtile(8)][ktile(2)][lane(32)][16]
__global__ void __launch_bounds__(256)
pack_x(const float* __restrict__ src, bf16_t* __restrict__ out) {
  int e = blockIdx.x * blockDim.x + threadIdx.x;
  if (e >= BB * SS * FF) return;
  int b = e / (SS * FF);
  int t = (e / FF) % SS;
  int k = e % FF;
  int mtile = b >> 4, row = b & 15;
  int ktile = k >> 5, kk = k & 31;
  int lane, h; amap(row, kk, lane, h);
  out[(size_t)t * (BB * FF) + (((size_t)mtile * KT_X + ktile) * 32 + lane) * 16 + h] =
      (bf16_t)src[e];
}

// zero c & hpack, out[:,0]=0, xcur = target[:,0,0]
__global__ void __launch_bounds__(256)
init_state(float* __restrict__ c, bf16_t* __restrict__ hpack,
           float* __restrict__ out, float* __restrict__ xcur,
           const float* __restrict__ target) {
  int e = blockIdx.x * blockDim.x + threadIdx.x;
  if (e < BB * HH) { c[e] = 0.0f; hpack[e] = (bf16_t)0.0f; }
  if (e < BB) { out[(size_t)e * TT] = 0.0f; xcur[e] = target[(size_t)e * TT]; }
}

// ================= per-step GEMM: gates(128x4096) = A(128xK) @ W^T ===========
// 8-wave (256-thread) blocks span 2 mtiles x 8 ntiles so that co-resident
// waves on the WGP share BOTH A fragments (x4) and B fragments (x2) through
// WGP$, cutting per-step L2 traffic ~2.4x vs a 1-mtile block. Each wave owns
// one mtile and two ntiles (A fragment reused across two WMMAs). Fragments
// are pre-packed: per-lane 32B contiguous -> 1KB/wave coalesced b128 loads.
__global__ void __launch_bounds__(256)
gemm_gates(const bf16_t* __restrict__ Ah,   // [mtile][32][32][16]  (hpack)
           const bf16_t* __restrict__ Ax,   // [mtile][ktx][32][16] or null
           const bf16_t* __restrict__ Bh,   // [ntile][32][32][16]
           const bf16_t* __restrict__ Bx,   // [ntile][ktx][32][16]
           float* __restrict__ gates, int ktx) {
  const int lane  = threadIdx.x & 31;
  const int wave  = threadIdx.x >> 5;                 // 0..7
  const int mtile = blockIdx.y * 2 + (wave & 1);      // 2 mtiles per block
  const int nt0   = blockIdx.x * 8 + (wave >> 1) * 2; // 8 ntiles per block

  v8f acc0 = {}, acc1 = {};

  const bf16_t* aB  = Ah + (size_t)mtile * KT_H * 512 + lane * 16;
  const bf16_t* bB0 = Bh + (size_t)(nt0 + 0) * KT_H * 512 + lane * 16;
  const bf16_t* bB1 = Bh + (size_t)(nt0 + 1) * KT_H * 512 + lane * 16;

#pragma unroll 2
  for (int kt = 0; kt < KT_H; ++kt) {
    bf16x16 a  = *(const bf16x16*)(aB  + (size_t)kt * 512);
    bf16x16 b0 = *(const bf16x16*)(bB0 + (size_t)kt * 512);
    bf16x16 b1 = *(const bf16x16*)(bB1 + (size_t)kt * 512);
    acc0 = __builtin_amdgcn_wmma_f32_16x16x32_bf16(false, a, false, b0, (short)0, acc0, false, false);
    acc1 = __builtin_amdgcn_wmma_f32_16x16x32_bf16(false, a, false, b1, (short)0, acc1, false, false);
  }

  if (Ax) {  // encoder: fold x_t @ Wih^T (2 extra k-tiles)
    const bf16_t* axB = Ax + (size_t)mtile * ktx * 512 + lane * 16;
#pragma unroll
    for (int kt = 0; kt < KT_X; ++kt) {
      bf16x16 a  = *(const bf16x16*)(axB + (size_t)kt * 512);
      bf16x16 b0 = *(const bf16x16*)(Bx + ((size_t)(nt0 + 0) * ktx + kt) * 512 + lane * 16);
      bf16x16 b1 = *(const bf16x16*)(Bx + ((size_t)(nt0 + 1) * ktx + kt) * 512 + lane * 16);
      acc0 = __builtin_amdgcn_wmma_f32_16x16x32_bf16(false, a, false, b0, (short)0, acc0, false, false);
      acc1 = __builtin_amdgcn_wmma_f32_16x16x32_bf16(false, a, false, b1, (short)0, acc1, false, false);
    }
  }

  // C/D layout (ISA 7.12.2): lane 0-15 -> N=lane, M=r ; lane 16-31 -> N=lane-16, M=8+r
  const int nlo   = lane & 15;
  const int rbase = (lane < 16) ? 0 : 8;
  float* g0 = gates + (size_t)(mtile * 16) * G4 + (nt0 + 0) * 16 + nlo;
  float* g1 = gates + (size_t)(mtile * 16) * G4 + (nt0 + 1) * 16 + nlo;
#pragma unroll
  for (int r = 0; r < 8; ++r) {
    g0[(size_t)(rbase + r) * G4] = acc0[r];
    g1[(size_t)(rbase + r) * G4] = acc1[r];
  }
}

// ================= encoder pointwise LSTM cell ==============================
__global__ void __launch_bounds__(256)
lstm_point_enc(const float* __restrict__ gates, const float* __restrict__ bih,
               const float* __restrict__ bhh, float* __restrict__ c,
               bf16_t* __restrict__ hpack) {
  int e = blockIdx.x * blockDim.x + threadIdx.x;  // B*H
  if (e >= BB * HH) return;
  int b = e >> 10, j = e & (HH - 1);
  const float* g = gates + (size_t)b * G4;
  float gi = g[j]          + bih[j]          + bhh[j];
  float gf = g[j + HH]     + bih[j + HH]     + bhh[j + HH];
  float gg = g[j + 2 * HH] + bih[j + 2 * HH] + bhh[j + 2 * HH];
  float go = g[j + 3 * HH] + bih[j + 3 * HH] + bhh[j + 3 * HH];
  float c2 = sigmf(gf) * c[e] + sigmf(gi) * tanhf(gg);
  c[e] = c2;
  float h2 = sigmf(go) * tanhf(c2);
  int mtile = b >> 4, row = b & 15, ktile = j >> 5, kk = j & 31;
  int lane, h; amap(row, kk, lane, h);
  hpack[(((size_t)mtile * KT_H + ktile) * 32 + lane) * 16 + h] = (bf16_t)h2;
}

// ====== decoder pointwise LSTM + fused fc head (one block per batch row) =====
__global__ void __launch_bounds__(256)
lstm_dec_step(const float* __restrict__ gates, const float* __restrict__ bih,
              const float* __restrict__ bhh, const float* __restrict__ wih,
              float* __restrict__ c, bf16_t* __restrict__ hpack,
              float* __restrict__ xcur, const float* __restrict__ fcW,
              const float* __restrict__ fcb, float* __restrict__ out, int tcol) {
  __shared__ float red[256];
  const int b = blockIdx.x;
  const float x = xcur[b];
  const float* g = gates + (size_t)b * G4;
  float partial = 0.0f;
#pragma unroll
  for (int q = 0; q < 4; ++q) {
    int j = threadIdx.x + q * 256;
    float gi = g[j]          + bih[j]          + bhh[j]          + x * wih[j];
    float gf = g[j + HH]     + bih[j + HH]     + bhh[j + HH]     + x * wih[j + HH];
    float gg = g[j + 2 * HH] + bih[j + 2 * HH] + bhh[j + 2 * HH] + x * wih[j + 2 * HH];
    float go = g[j + 3 * HH] + bih[j + 3 * HH] + bhh[j + 3 * HH] + x * wih[j + 3 * HH];
    float c2 = sigmf(gf) * c[(size_t)b * HH + j] + sigmf(gi) * tanhf(gg);
    c[(size_t)b * HH + j] = c2;
    float h2 = sigmf(go) * tanhf(c2);
    int mtile = b >> 4, row = b & 15, ktile = j >> 5, kk = j & 31;
    int lane, h; amap(row, kk, lane, h);
    hpack[(((size_t)mtile * KT_H + ktile) * 32 + lane) * 16 + h] = (bf16_t)h2;
    partial += h2 * fcW[j];
  }
  red[threadIdx.x] = partial;
  __syncthreads();
  for (int s = 128; s > 0; s >>= 1) {
    if (threadIdx.x < s) red[threadIdx.x] += red[threadIdx.x + s];
    __syncthreads();
  }
  if (threadIdx.x == 0) {
    float pred = red[0] + fcb[0];
    out[(size_t)b * TT + tcol] = pred;
    xcur[b] = pred;
  }
}

// =============================== host side ==================================
extern "C" void kernel_launch(void* const* d_in, const int* in_sizes, int n_in,
                              void* d_out, int out_size, void* d_ws, size_t ws_size,
                              hipStream_t stream) {
  const float* source  = (const float*)d_in[0];
  const float* target  = (const float*)d_in[1];
  const float* encWih  = (const float*)d_in[2];
  const float* encWhh  = (const float*)d_in[3];
  const float* encBih  = (const float*)d_in[4];
  const float* encBhh  = (const float*)d_in[5];
  const float* decWih  = (const float*)d_in[6];
  const float* decWhh  = (const float*)d_in[7];
  const float* decBih  = (const float*)d_in[8];
  const float* decBhh  = (const float*)d_in[9];
  const float* fcW     = (const float*)d_in[10];
  const float* fcb     = (const float*)d_in[11];
  float* out = (float*)d_out;

  char* ws = (char*)d_ws;
  size_t off = 0;
  auto alloc = [&](size_t bytes) { char* p = ws + off; off = (off + bytes + 255) & ~(size_t)255; return p; };
  bf16_t* BhEnc = (bf16_t*)alloc((size_t)G4 * HH * 2);      // 8 MB
  bf16_t* BxEnc = (bf16_t*)alloc((size_t)G4 * FF * 2);      // 0.5 MB
  bf16_t* BhDec = (bf16_t*)alloc((size_t)G4 * HH * 2);      // 8 MB
  bf16_t* AxAll = (bf16_t*)alloc((size_t)BB * SS * FF * 2); // 8 MB
  bf16_t* hpack = (bf16_t*)alloc((size_t)BB * HH * 2);
  float*  gates = (float*)alloc((size_t)BB * G4 * 4);
  float*  cbuf  = (float*)alloc((size_t)BB * HH * 4);
  float*  xcur  = (float*)alloc((size_t)BB * 4);

  // one-time packs (parallel, cheap vs 639 sequential GEMMs)
  pack_B<<<(G4 * HH + 255) / 256, 256, 0, stream>>>(encWhh, BhEnc, HH, KT_H);
  pack_B<<<(G4 * FF + 255) / 256, 256, 0, stream>>>(encWih, BxEnc, FF, KT_X);
  pack_B<<<(G4 * HH + 255) / 256, 256, 0, stream>>>(decWhh, BhDec, HH, KT_H);
  pack_x<<<(BB * SS * FF + 255) / 256, 256, 0, stream>>>(source, AxAll);
  init_state<<<(BB * HH + 255) / 256, 256, 0, stream>>>(cbuf, hpack, out, xcur, target);

  // 256 ntiles / 8-per-block -> grid.x = 32 ; 8 mtiles / 2-per-block -> grid.y = 4
  dim3 ggrid(32, 4);
  // ---------------- encoder: 512 sequential steps ----------------
  for (int t = 0; t < SS; ++t) {
    gemm_gates<<<ggrid, 256, 0, stream>>>(hpack, AxAll + (size_t)t * BB * FF,
                                          BhEnc, BxEnc, gates, KT_X);
    lstm_point_enc<<<(BB * HH + 255) / 256, 256, 0, stream>>>(gates, encBih, encBhh,
                                                              cbuf, hpack);
  }
  // ---------------- decoder: 127 sequential steps ----------------
  for (int t = 0; t < TT - 1; ++t) {
    gemm_gates<<<ggrid, 256, 0, stream>>>(hpack, nullptr, BhDec, nullptr, gates, 0);
    lstm_dec_step<<<BB, 256, 0, stream>>>(gates, decBih, decBhh, decWih,
                                          cbuf, hpack, xcur, fcW, fcb, out, t + 1);
  }
}
